// LRGCNModel_49529562857562
// MI455X (gfx1250) — matrix operand order repaired and checked by reading, not compiled, not run on hardware
//
#include <hip/hip_runtime.h>
#include <hip/hip_bf16.h>

#define NN 50000
#define NE 800000
#define KD 512      // concat K: [agg_x | x | agg_h | h], each 128

typedef __attribute__((ext_vector_type(16))) __bf16 v16bf;
typedef __attribute__((ext_vector_type(8)))  float  v8f;

union FragBF { v16bf v; uint4 q[2]; };

__device__ __forceinline__ unsigned int f2bf(float f) {
    unsigned int u = __builtin_bit_cast(unsigned int, f);
    u += 0x7FFFu + ((u >> 16) & 1u);   // round-to-nearest-even
    return u >> 16;
}

// ---------------- zero accumulators ----------------
__global__ void k_zero(float* __restrict__ p, int n) {
    int i = blockIdx.x * 256 + threadIdx.x;
    if (i < n) p[i] = 0.0f;
}

// ---------------- edge scatter-sum (64 threads per edge) ----------------
__global__ void k_scatter(const float* __restrict__ x, const float* __restrict__ h,
                          const int* __restrict__ ei,
                          float* __restrict__ sum_x, float* __restrict__ sum_h,
                          float* __restrict__ cnt) {
    int gid  = blockIdx.x * 256 + threadIdx.x;       // < NE*64 exactly
    int e    = gid >> 6;
    int slot = gid & 63;
    int src = ei[e];
    int dst = ei[NE + e];
    if (slot < 32) {
        int f = slot * 4;
        const float4 v = *(const float4*)(x + (size_t)src * 128 + f);
        float* p = sum_x + (size_t)dst * 128 + f;
        atomicAdd(p + 0, v.x); atomicAdd(p + 1, v.y);
        atomicAdd(p + 2, v.z); atomicAdd(p + 3, v.w);
        if (slot == 0) atomicAdd(cnt + dst, 1.0f);
    } else {
        int f = (slot - 32) * 4;
        const float4 v = *(const float4*)(h + (size_t)src * 128 + f);
        float* p = sum_h + (size_t)dst * 128 + f;
        atomicAdd(p + 0, v.x); atomicAdd(p + 1, v.y);
        atomicAdd(p + 2, v.z); atomicAdd(p + 3, v.w);
    }
}

// ---------------- normalize + pack A = [agg_x | x | agg_h | h] as bf16 ----------------
__global__ void k_pack(const float* __restrict__ x, const float* __restrict__ h,
                       const float* __restrict__ sum_x, const float* __restrict__ sum_h,
                       const float* __restrict__ cnt, unsigned short* __restrict__ A) {
    int gid = blockIdx.x * 256 + threadIdx.x;        // < NN*128 exactly
    int n = gid >> 7;
    int k0 = (gid & 127) * 4;
    float inv = 1.0f / fmaxf(cnt[n], 1.0f);
    float4 v;
    if (k0 < 128) {
        v = *(const float4*)(sum_x + (size_t)n * 128 + k0);
        v.x *= inv; v.y *= inv; v.z *= inv; v.w *= inv;
    } else if (k0 < 256) {
        v = *(const float4*)(x + (size_t)n * 128 + (k0 - 128));
    } else if (k0 < 384) {
        v = *(const float4*)(sum_h + (size_t)n * 128 + (k0 - 256));
        v.x *= inv; v.y *= inv; v.z *= inv; v.w *= inv;
    } else {
        v = *(const float4*)(h + (size_t)n * 128 + (k0 - 384));
    }
    uint2 o;
    o.x = f2bf(v.x) | (f2bf(v.y) << 16);
    o.y = f2bf(v.z) | (f2bf(v.w) << 16);
    *(uint2*)(A + (size_t)n * KD + k0) = o;
}

// ---------------- stack + transpose weights: Wt[gate][col][k] (bf16) ----------------
__global__ void k_wt(const float* __restrict__ Wx, const float* __restrict__ Wxr,
                     const float* __restrict__ Wh, const float* __restrict__ Whr,
                     unsigned short* __restrict__ Wt) {
    int gid = blockIdx.x * 256 + threadIdx.x;        // < 4*128*512 exactly
    int gate = gid >> 16;
    int rem  = gid & 65535;
    int col  = rem >> 9;
    int k    = rem & 511;
    float v;
    if (k < 128)      v = Wx [((size_t)gate * 128 + k        ) * 128 + col];
    else if (k < 256) v = Wxr[((size_t)gate * 128 + (k - 128)) * 128 + col];
    else if (k < 384) v = Wh [((size_t)gate * 128 + (k - 256)) * 128 + col];
    else              v = Whr[((size_t)gate * 128 + (k - 384)) * 128 + col];
    Wt[((size_t)gate * 128 + col) * KD + k] = (unsigned short)f2bf(v);
}

// ---------------- fused: WMMA GEMM (K=512, 4 gates) + LSTM + y ----------------
__global__ __launch_bounds__(256) void k_fused(
        const unsigned short* __restrict__ A, const unsigned short* __restrict__ Wt,
        const float* __restrict__ bx, const float* __restrict__ bh,
        const float* __restrict__ c_in, const float* __restrict__ fc_w,
        const float* __restrict__ fc_b,
        float* __restrict__ y_out, float* __restrict__ h_out, float* __restrict__ c_out) {
    __shared__ unsigned short Alds[16 * 520];   // 16 rows x 512 bf16, pad->520 (bank spread)
    __shared__ float glds[4 * 16 * 132];        // g[gate][row][col], pad 128->132

    const int tid   = threadIdx.x;
    const int node0 = blockIdx.x * 16;

    // Stage A tile (16 x 512 bf16 = 16 KB) with gfx1250 async global->LDS copies:
    // each lane supplies its own LDS destination (padded-row remap), 4 x b128 per thread.
    #pragma unroll
    for (int i = 0; i < 4; ++i) {
        int idx8 = tid + 256 * i;                              // 8-short chunk id
        int row = idx8 >> 6;
        int kk  = (idx8 & 63) * 8;
        unsigned lds_off = (unsigned)(unsigned long long)(void*)&Alds[row * 520 + kk];
        const void* gp = (const void*)(A + (size_t)node0 * KD + (size_t)idx8 * 8);
        asm volatile("global_load_async_to_lds_b128 %0, %1, off"
                     :: "v"(lds_off), "v"(gp) : "memory");
    }
    asm volatile("s_wait_asynccnt 0" ::: "memory");
    __syncthreads();

    const int wave = tid >> 5, lane = tid & 31;
    const int half = lane >> 4, lrow = lane & 15;

    // Each of 8 waves computes 4 of the 32 (gate, 16-col-tile) output tiles.
    #pragma unroll
    for (int t4 = 0; t4 < 4; ++t4) {
        int task  = wave * 4 + t4;
        int gate  = task >> 3;
        int ctile = task & 7;
        int col   = ctile * 16 + lrow;
        const unsigned short* bp = Wt + ((size_t)gate * 128 + col) * KD + half * 8;
        const unsigned short* ap = &Alds[lrow * 520 + half * 8];
        __builtin_prefetch(bp, 0, 3);   // global_prefetch_b8, WGP scope (near caches)
        v8f acc = {};
        #pragma unroll
        for (int kk = 0; kk < 16; ++kk) {
            FragBF a, b;
            a.q[0] = *(const uint4*)(ap + kk * 32);        // K = kb + half*8 .. +7
            a.q[1] = *(const uint4*)(ap + kk * 32 + 16);   // K = kb+16+half*8 .. +7
            b.q[0] = *(const uint4*)(bp + kk * 32);        // K = kb + half*8 .. +7
            b.q[1] = *(const uint4*)(bp + kk * 32 + 16);   // K = kb+16+half*8 .. +7
            acc = __builtin_amdgcn_wmma_f32_16x16x32_bf16(
                      false, a.v, false, b.v, (short)0, acc, false, false);
        }
        float bias = bx[gate * 128 + col] + bh[gate * 128 + col];
        #pragma unroll
        for (int v = 0; v < 8; ++v) {
            glds[(gate * 16 + (v + half * 8)) * 132 + col] = acc[v] + bias;
        }
    }
    __syncthreads();

    // LSTM gates + output head: 16 threads per node, 8 dims each.
    const int node = tid >> 4, sub = tid & 15;
    const int gn = node0 + node;
    float ysum = 0.0f;
    #pragma unroll
    for (int j = 0; j < 8; ++j) {
        int d  = sub + j * 16;
        float gi = glds[(0 * 16 + node) * 132 + d];
        float gf = glds[(1 * 16 + node) * 132 + d];
        float gc = glds[(2 * 16 + node) * 132 + d];
        float go = glds[(3 * 16 + node) * 132 + d];
        float ig = 1.0f / (1.0f + expf(-gi));
        float fg = 1.0f / (1.0f + expf(-gf));
        float tg = tanhf(gc);
        float og = 1.0f / (1.0f + expf(-go));
        float cn = fg * c_in[(size_t)gn * 128 + d] + ig * tg;
        float hn = og * tanhf(cn);
        h_out[(size_t)gn * 128 + d] = hn;
        c_out[(size_t)gn * 128 + d] = cn;
        ysum += fmaxf(hn, 0.0f) * fc_w[d];
    }
    ysum += __shfl_xor(ysum, 1, 16);
    ysum += __shfl_xor(ysum, 2, 16);
    ysum += __shfl_xor(ysum, 4, 16);
    ysum += __shfl_xor(ysum, 8, 16);
    if (sub == 0) y_out[gn] = ysum + fc_b[0];
}

extern "C" void kernel_launch(void* const* d_in, const int* in_sizes, int n_in,
                              void* d_out, int out_size, void* d_ws, size_t ws_size,
                              hipStream_t stream) {
    (void)in_sizes; (void)n_in; (void)out_size; (void)ws_size;
    const float* x   = (const float*)d_in[0];
    const int*   ei  = (const int*)d_in[1];
    // d_in[2] edge_weight: single relation, unused
    const float* h   = (const float*)d_in[3];
    const float* c   = (const float*)d_in[4];
    const float* Wx  = (const float*)d_in[5];
    const float* Wxr = (const float*)d_in[6];
    const float* bx  = (const float*)d_in[7];
    const float* Wh  = (const float*)d_in[8];
    const float* Whr = (const float*)d_in[9];
    const float* bh  = (const float*)d_in[10];
    const float* fcw = (const float*)d_in[11];
    const float* fcb = (const float*)d_in[12];

    char* ws = (char*)d_ws;
    float* sum_x = (float*)ws;                               // NN*128 f32
    float* sum_h = sum_x + (size_t)NN * 128;                 // NN*128 f32
    float* cnt   = sum_h + (size_t)NN * 128;                 // NN f32
    unsigned short* Abuf = (unsigned short*)(ws + 51400192); // NN*512 bf16 (16B aligned)
    unsigned short* Wt   = (unsigned short*)(ws + 51400192 + (size_t)NN * KD * 2);

    float* y_out = (float*)d_out;                            // [NN]
    float* h_out = y_out + NN;                               // [NN,128]
    float* c_out = h_out + (size_t)NN * 128;                 // [NN,128]

    const int nz = NN * 257;                                 // sum_x + sum_h + cnt
    k_zero   <<<(nz + 255) / 256, 256, 0, stream>>>(sum_x, nz);
    k_scatter<<<(NE * 64) / 256, 256, 0, stream>>>(x, h, ei, sum_x, sum_h, cnt);
    k_pack   <<<(NN * 128) / 256, 256, 0, stream>>>(x, h, sum_x, sum_h, cnt, Abuf);
    k_wt     <<<(4 * 128 * 512) / 256, 256, 0, stream>>>(Wx, Wxr, Wh, Whr, Wt);
    k_fused  <<<NN / 16, 256, 0, stream>>>(Abuf, Wt, bx, bh, c, fcw, fcb,
                                           y_out, h_out, c_out);
}
